// Attention_24421184045466
// MI455X (gfx1250) — matrix operand order repaired and checked by reading, not compiled, not run on hardware
//
#include <hip/hip_runtime.h>
#include <hip/hip_bf16.h>

// ---------------------------------------------------------------------------
// Coverage-attention (pointer-generator) for MI455X / gfx1250, wave32 + WMMA.
// B=32, S=4096, H2=512.
// ---------------------------------------------------------------------------

#define BB    32
#define SLEN  4096
#define H2    512
#define TM    64          // S-rows per scores block
#define LPITCH 520        // bf16 elements per LDS row (512 + 8 pad -> conflict-free b128)

typedef __attribute__((ext_vector_type(16))) __bf16 v16bf;
typedef __attribute__((ext_vector_type(8)))  __bf16 v8bf;
typedef __attribute__((ext_vector_type(8)))  float  v8f;

__device__ __forceinline__ unsigned short f32_to_bf16(float f) {
    unsigned int u = __float_as_uint(f);
    unsigned int r = u + 0x7FFFu + ((u >> 16) & 1u);   // round-to-nearest-even
    return (unsigned short)(r >> 16);
}

__device__ __forceinline__ v16bf cat8(v8bf lo, v8bf hi) {
    return __builtin_shufflevector(lo, hi, 0,1,2,3,4,5,6,7,8,9,10,11,12,13,14,15);
}

__device__ __forceinline__ v8f wmma_bf16(v16bf a, v16bf b, v8f c) {
    // (neg_a, A, neg_b, B, c_mod, C, reuse_a, reuse_b)
    return __builtin_amdgcn_wmma_f32_16x16x32_bf16(false, a, false, b, (short)0, c,
                                                   false, false);
}

// ---------------------------------------------------------------------------
// Kernel 1: d[b,n] = dec[b,:]@W_s[:,n] + b_h[n] + b_s[n] + b_c[n]     [32,512]
// ---------------------------------------------------------------------------
__global__ void dvec_kernel(const float* __restrict__ dec,
                            const float* __restrict__ W_s,
                            const float* __restrict__ b_h,
                            const float* __restrict__ b_s,
                            const float* __restrict__ b_c,
                            float* __restrict__ dvec) {
    const int gid = blockIdx.x * 256 + threadIdx.x;   // 64 blocks
    const int b = gid >> 9;
    const int n = gid & (H2 - 1);
    float acc = b_h[n] + b_s[n] + b_c[n];
    const float* db = dec + b * H2;
    for (int k = 0; k < H2; ++k)
        acc += db[k] * W_s[(size_t)k * H2 + n];       // db[k] scalar, W_s coalesced
    dvec[b * H2 + n] = acc;
}

// ---------------------------------------------------------------------------
// Kernel 2: pack W_h / W_c (fp32 row-major [K,N]) into bf16 WMMA B-fragment
// layout: Wpack[((j*16+kt)*32 + lane)*16 + e]  where h=lane>>4, N=j*16+(lane&15),
// K=kt*32+8h+(e&7)+16*(e>>3) (ISA table order).  Each lane's 16
// elements are contiguous (32B) -> 1KB fully-coalesced fragment loads.
// ---------------------------------------------------------------------------
__global__ void packW_kernel(const float* __restrict__ W_h,
                             const float* __restrict__ W_c,
                             unsigned short* __restrict__ wph,
                             unsigned short* __restrict__ wpc) {
    const int flat = blockIdx.x * 256 + threadIdx.x;  // 2048 blocks => 524288
    const int mat  = flat >> 18;                      // 262144 elems per matrix
    const int idx  = flat & 262143;
    const int j    = idx >> 13;                       // 8192 per n-tile
    const int rem  = idx & 8191;
    const int kt   = rem >> 9;                        // 512 per k-tile
    const int rem2 = rem & 511;
    const int lane = rem2 >> 4;
    const int e    = rem2 & 15;
    const int n    = j * 16 + (lane & 15);
    const int k    = kt * 32 + (lane >> 4) * 8 + (e & 7) + (e >> 3) * 16;
    const float* src = mat ? W_c : W_h;
    unsigned short* dst = mat ? wpc : wph;
    dst[idx] = f32_to_bf16(src[(size_t)k * H2 + n]);
}

// ---------------------------------------------------------------------------
// Kernel 3: scores[b,s] = sum_n tanh( enc@W_h + cov@W_c + d[b,n] ) * v[n]
// WMMA GEMM, M=B*S, N=K=512, fused tanh/dot-v epilogue.
// grid = 32*64 blocks, 256 threads (8 wave32). ~135KB dynamic LDS.
// Inner k-step: batch all 8 A fragments into distinct registers, then issue
// 8 back-to-back WMMAs (dependent accumulator pairs separated by 3 others).
// ---------------------------------------------------------------------------
__global__ void __launch_bounds__(256, 2)
scores_kernel(const float* __restrict__ enc,
              const float* __restrict__ cov,
              const unsigned short* __restrict__ wph,
              const unsigned short* __restrict__ wpc,
              const float* __restrict__ dvec,
              const float* __restrict__ v,
              float* __restrict__ scores) {
    extern __shared__ __align__(16) char smem_raw[];
    unsigned short* Ae = (unsigned short*)smem_raw;          // [TM][LPITCH] bf16
    unsigned short* Ac = Ae + TM * LPITCH;
    float* spart = (float*)(Ac + TM * LPITCH);               // [8][64]

    const int b     = blockIdx.x >> 6;
    const int stile = blockIdx.x & 63;
    const int row0  = stile * TM;
    const int t     = threadIdx.x;

    // ---- stage enc/cov tile (64x512 fp32, contiguous) as bf16 into LDS ----
    const float4* encB = (const float4*)(enc + ((size_t)b * SLEN + row0) * H2);
    const float4* covB = (const float4*)(cov + ((size_t)b * SLEN + row0) * H2);
#pragma unroll
    for (int i = 0; i < 32; ++i) {
        const int vi = t + 256 * i;            // 8192 float4 per matrix
        const int r  = vi >> 7;
        const int c4 = vi & 127;
        {
            float4 x = encB[vi];
            unsigned int lo = (unsigned int)f32_to_bf16(x.x) |
                              ((unsigned int)f32_to_bf16(x.y) << 16);
            unsigned int hi = (unsigned int)f32_to_bf16(x.z) |
                              ((unsigned int)f32_to_bf16(x.w) << 16);
            *(uint2*)(Ae + r * LPITCH + c4 * 4) = make_uint2(lo, hi);
        }
        {
            float4 x = covB[vi];
            unsigned int lo = (unsigned int)f32_to_bf16(x.x) |
                              ((unsigned int)f32_to_bf16(x.y) << 16);
            unsigned int hi = (unsigned int)f32_to_bf16(x.z) |
                              ((unsigned int)f32_to_bf16(x.w) << 16);
            *(uint2*)(Ac + r * LPITCH + c4 * 4) = make_uint2(lo, hi);
        }
    }
    __syncthreads();

    const int w       = t >> 5;          // wave 0..7 -> owns n-tiles j = w + 8*jj
    const int lane    = t & 31;
    const int halfsel = lane >> 4;       // A-frag K-half select per ISA layout
    const int lrow    = lane & 15;

    const v16bf* wph16 = (const v16bf*)wph;
    const v16bf* wpc16 = (const v16bf*)wpc;

    float sacc[4][8];
#pragma unroll
    for (int mt = 0; mt < 4; ++mt)
#pragma unroll
        for (int i = 0; i < 8; ++i) sacc[mt][i] = 0.0f;

    const v8f zero = {0.f, 0.f, 0.f, 0.f, 0.f, 0.f, 0.f, 0.f};

    for (int jj = 0; jj < 4; ++jj) {
        const int j = w + jj * 8;        // n-tile 0..31
        v8f c[4] = {zero, zero, zero, zero};

#pragma unroll 2
        for (int kt = 0; kt < 16; ++kt) {
            // B fragments: one 32B contiguous read per lane, reused for 4 M-tiles
            const v16bf bh = wph16[(j * 16 + kt) * 32 + lane];
            const v16bf bc = wpc16[(j * 16 + kt) * 32 + lane];
            const int kb = kt * 32 + halfsel * 8;

            // batch-load all 8 A fragments into distinct registers (clauseable)
            v16bf ae[4], ac[4];
#pragma unroll
            for (int mt = 0; mt < 4; ++mt) {
                const unsigned short* ap = Ae + (mt * 16 + lrow) * LPITCH + kb;
                ae[mt] = cat8(*(const v8bf*)ap, *(const v8bf*)(ap + 16));
                const unsigned short* cp = Ac + (mt * 16 + lrow) * LPITCH + kb;
                ac[mt] = cat8(*(const v8bf*)cp, *(const v8bf*)(cp + 16));
            }
            // 8 back-to-back WMMAs; each c[mt] RAW pair is 4 instructions apart
#pragma unroll
            for (int mt = 0; mt < 4; ++mt)
                c[mt] = wmma_bf16(ae[mt], bh, c[mt]);
#pragma unroll
            for (int mt = 0; mt < 4; ++mt)
                c[mt] = wmma_bf16(ac[mt], bc, c[mt]);
        }
        // fused epilogue: + d[b,n], tanh, * v[n], reduce over N into row partials
        const int n  = j * 16 + lrow;
        const float dn = dvec[b * H2 + n];
        const float vn = v[n];
#pragma unroll
        for (int mt = 0; mt < 4; ++mt)
#pragma unroll
            for (int i = 0; i < 8; ++i)
                sacc[mt][i] += tanhf(c[mt][i] + dn) * vn;
    }

    // reduce across the 16 lanes of each C half (rows stay fixed under xor<=8)
#pragma unroll
    for (int mt = 0; mt < 4; ++mt)
#pragma unroll
        for (int i = 0; i < 8; ++i) {
            float s = sacc[mt][i];
            s += __shfl_xor(s, 1);
            s += __shfl_xor(s, 2);
            s += __shfl_xor(s, 4);
            s += __shfl_xor(s, 8);
            if (lrow == 0)
                spart[w * 64 + mt * 16 + halfsel * 8 + i] = s;
        }
    __syncthreads();

    if (t < TM) {
        float s = 0.0f;
#pragma unroll
        for (int ww = 0; ww < 8; ++ww) s += spart[ww * 64 + t];
        scores[b * SLEN + row0 + t] = s;
    }
}

// ---------------------------------------------------------------------------
// Kernel 4: softmax over S per batch. 32 blocks x 256 threads, 16 vals/thread.
// ---------------------------------------------------------------------------
__global__ void softmax_kernel(const float* __restrict__ scores,
                               float* __restrict__ attn) {
    const int b = blockIdx.x;
    const int t = threadIdx.x;
    __shared__ float red[8];
    __shared__ float bval;
    const float* sb = scores + b * SLEN;

    float vals[16];
    float m = -3.402823466e38f;
#pragma unroll
    for (int k = 0; k < 16; ++k) {
        vals[k] = sb[t + 256 * k];
        m = fmaxf(m, vals[k]);
    }
    for (int x = 1; x < 32; x <<= 1) m = fmaxf(m, __shfl_xor(m, x));
    if ((t & 31) == 0) red[t >> 5] = m;
    __syncthreads();
    if (t == 0) {
        float mm = red[0];
        for (int i = 1; i < 8; ++i) mm = fmaxf(mm, red[i]);
        bval = mm;
    }
    __syncthreads();
    const float M = bval;

    float s = 0.0f;
#pragma unroll
    for (int k = 0; k < 16; ++k) {
        vals[k] = __expf(vals[k] - M);
        s += vals[k];
    }
    for (int x = 1; x < 32; x <<= 1) s += __shfl_xor(s, x);
    __syncthreads();
    if ((t & 31) == 0) red[t >> 5] = s;
    __syncthreads();
    if (t == 0) {
        float ss = 0.0f;
        for (int i = 0; i < 8; ++i) ss += red[i];
        bval = ss;
    }
    __syncthreads();
    const float inv = 1.0f / bval;
#pragma unroll
    for (int k = 0; k < 16; ++k)
        attn[b * SLEN + t + 256 * k] = vals[k] * inv;
}

// ---------------------------------------------------------------------------
// Kernel 5: fused streaming pass:
//   coverage_new = cov + attn (broadcast), partial h_star = sum_s attn*enc.
// grid = 32 b x 16 chunks (256 rows each), 256 threads (2 cols/thread).
// ---------------------------------------------------------------------------
__global__ void ctx_cov_kernel(const float* __restrict__ enc,
                               const float* __restrict__ cov,
                               const float* __restrict__ attn,
                               float* __restrict__ covnew,
                               float* __restrict__ hpart) {
    const int b  = blockIdx.x >> 4;
    const int ch = blockIdx.x & 15;
    const int t  = threadIdx.x;
    const int c0 = t;
    const int c1 = t + 256;
    float h0 = 0.0f, h1 = 0.0f;
    const size_t base = ((size_t)b * SLEN + ch * 256) * H2;
    const float* ab = attn + b * SLEN + ch * 256;
    for (int r = 0; r < 256; ++r) {
        const float a = ab[r];                 // uniform across threads -> scalar
        const size_t off = base + (size_t)r * H2;
        const float e0 = enc[off + c0];
        const float e1 = enc[off + c1];
        covnew[off + c0] = cov[off + c0] + a;
        covnew[off + c1] = cov[off + c1] + a;
        h0 += a * e0;
        h1 += a * e1;
    }
    hpart[((size_t)(b * 16 + ch)) * H2 + c0] = h0;
    hpart[((size_t)(b * 16 + ch)) * H2 + c1] = h1;
}

// ---------------------------------------------------------------------------
// Kernel 6: deterministic reduction of chunk partials -> h_star [32,512]
// ---------------------------------------------------------------------------
__global__ void hstar_reduce_kernel(const float* __restrict__ hpart,
                                    float* __restrict__ hstar) {
    const int gid = blockIdx.x * 256 + threadIdx.x;   // 64 blocks
    const int b = gid >> 9;
    const int c = gid & (H2 - 1);
    float s = 0.0f;
#pragma unroll
    for (int ch = 0; ch < 16; ++ch)
        s += hpart[((size_t)(b * 16 + ch)) * H2 + c];
    hstar[b * H2 + c] = s;
}

// ---------------------------------------------------------------------------
// Launch
// ---------------------------------------------------------------------------
extern "C" void kernel_launch(void* const* d_in, const int* in_sizes, int n_in,
                              void* d_out, int out_size, void* d_ws, size_t ws_size,
                              hipStream_t stream) {
    (void)in_sizes; (void)n_in; (void)out_size; (void)ws_size;

    const float* dec  = (const float*)d_in[0];   // [32,1,512]
    const float* enc  = (const float*)d_in[1];   // [32,4096,512]
    const float* cov  = (const float*)d_in[2];   // [32,4096,512]
    const float* W_h  = (const float*)d_in[3];
    const float* b_h  = (const float*)d_in[4];
    const float* W_s  = (const float*)d_in[5];
    const float* b_s  = (const float*)d_in[6];
    const float* W_c  = (const float*)d_in[7];
    const float* b_c  = (const float*)d_in[8];
    const float* v    = (const float*)d_in[9];   // [512,1]

    // d_out layout: h_star [32*512] | attn [32*4096] | coverage_new [32*4096*512]
    float* out_h = (float*)d_out;
    float* out_a = out_h + BB * H2;
    float* out_c = out_a + BB * SLEN;

    // workspace layout (bytes)
    char* ws = (char*)d_ws;
    float*          dvec   = (float*)(ws + 0);                 //  64 KB
    float*          scores = (float*)(ws + 65536);             // 512 KB
    float*          hpart  = (float*)(ws + 589824);            //   1 MB
    unsigned short* wph    = (unsigned short*)(ws + 1638400);  // 512 KB
    unsigned short* wpc    = (unsigned short*)(ws + 2162688);  // 512 KB

    dvec_kernel<<<64, 256, 0, stream>>>(dec, W_s, b_h, b_s, b_c, dvec);
    packW_kernel<<<2048, 256, 0, stream>>>(W_h, W_c, wph, wpc);

    const size_t smem = (size_t)(2 * TM * LPITCH) * sizeof(unsigned short)
                        + 8 * 64 * sizeof(float);              // ~135 KB
    scores_kernel<<<BB * (SLEN / TM), 256, smem, stream>>>(enc, cov, wph, wpc,
                                                           dvec, v, scores);

    softmax_kernel<<<BB, 256, 0, stream>>>(scores, out_a);
    ctx_cov_kernel<<<BB * 16, 256, 0, stream>>>(enc, cov, out_a, out_c, hpart);
    hstar_reduce_kernel<<<64, 256, 0, stream>>>(hpart, out_h);
}
